// MultiHeadAttention_13967233647646
// MI455X (gfx1250) — matrix operand order repaired and checked by reading, not compiled, not run on hardware
//
#include <hip/hip_runtime.h>
#include <hip/hip_bf16.h>

typedef __bf16 bf16_t;
typedef __attribute__((ext_vector_type(16))) __bf16 v16bf;
typedef __attribute__((ext_vector_type(8)))  float  v8f;
typedef __attribute__((ext_vector_type(4)))  unsigned int u32x4;
typedef __attribute__((ext_vector_type(8)))  int i32x8;
typedef __attribute__((ext_vector_type(4)))  int i32x4;

constexpr int kH  = 1024;    // hidden
constexpr int kNH = 16;      // heads
constexpr int kHD = 64;      // head dim
constexpr int kB  = 2;       // batch
constexpr int kS  = 2048;    // seq
constexpr int kM  = kB * kS; // 4096 rows

union FragU { v16bf v; uint4 q[2]; };

// A-operand (16 x K=32 window): lane = row, halves 0..7 -> k = hi*8..+7,
// halves 8..15 -> k = 16+hi*8..+7 (hi = lane>=16). ISA 16-bit A layout.
__device__ __forceinline__ v16bf load_fragA_s(const bf16_t* base, int lane, int stride) {
  const int r = lane & 15, hi = lane >> 4;
  const bf16_t* p = base + r * stride + hi * 8;
  FragU f;
  f.q[0] = *(const uint4*)(p);
  f.q[1] = *(const uint4*)(p + 16);
  return f.v;
}

// B-operand (K=32 x 16 window) from [n][k] storage: lane = col n, halves 0..7
// -> k = hb*8..+7, halves 8..15 -> k = 16+hb*8..+7 (hb = lane>=16). ISA 16-bit B layout.
__device__ __forceinline__ v16bf load_fragB_s(const bf16_t* base, int lane, int stride) {
  const int n = lane & 15, hb = lane >> 4;
  const bf16_t* p = base + n * stride + hb * 8;
  FragU f;
  f.q[0] = *(const uint4*)(p);
  f.q[1] = *(const uint4*)(p + 16);
  return f.v;
}

// ----------------------------------------------------------- TDM 2-D tile load
// Builds a Tensor DMA Descriptor (D#) per CDNA5 ISA and
// issues TENSOR_LOAD_TO_LDS via the 6-arg clang builtin (groups 0-3 + extra
// group + cpol). 2-D tile: tile_d1 rows x tile_d0 elements of 2-byte data,
// source row stride = row_stride elements, packed row-major into LDS at
// lds_byte_addr. Tracked by TENSORcnt.
__device__ __forceinline__ void tdm_load_tile_2d_bf16(
    unsigned int lds_byte_addr, const bf16_t* gptr,
    unsigned int tile_d0, unsigned int tile_d1, unsigned long long row_stride) {
  unsigned long long ga = (unsigned long long)(uintptr_t)gptr;
  u32x4 g0;
  g0[0] = 1u;                                   // count=1, user descriptor
  g0[1] = lds_byte_addr;                        // D#.lds_addr
  g0[2] = (unsigned int)ga;                     // global_addr[31:0]
  g0[3] = (unsigned int)((ga >> 32) & 0x1FFFFFFu) | (2u << 30); // addr[56:32], type=2
  i32x8 g1;
  g1[0] = (int)(1u << 16);                      // workgroup_mask=0, data_size=1 (2B)
  g1[1] = (int)((tile_d0 & 0xFFFFu) << 16);     // tensor_dim0[15:0] (== tile_d0)
  g1[2] = (int)((tile_d0 >> 16) | ((tile_d1 & 0xFFFFu) << 16)); // td0 hi | tensor_dim1 lo
  g1[3] = (int)((tile_d1 >> 16) | (tile_d0 << 16));             // td1 hi | tile_dim0
  g1[4] = (int)(tile_d1 & 0xFFFFu);             // tile_dim1, tile_dim2=0
  g1[5] = (int)(unsigned int)(row_stride & 0xFFFFFFFFu);        // dim0_stride lo
  g1[6] = (int)(unsigned int)((row_stride >> 32) & 0xFFFFu);    // dim0_stride hi
  g1[7] = 0;                                    // dim1_stride (unused, 2-D)
  i32x4 z4 = {0, 0, 0, 0};
  i32x8 z8 = {0, 0, 0, 0, 0, 0, 0, 0};
  __builtin_amdgcn_tensor_load_to_lds(g0, g1, z4, z4, z8, 0);
}

// ---------------------------------------------------------------- f32 -> bf16
__global__ __launch_bounds__(256) void cvt_f32_bf16_kernel(
    const float* __restrict__ src, bf16_t* __restrict__ dst, int n) {
  int i = (blockIdx.x * 256 + threadIdx.x) * 4;
  if (i + 3 < n) {
    float4 f = *(const float4*)(src + i);
    dst[i + 0] = (bf16_t)f.x;
    dst[i + 1] = (bf16_t)f.y;
    dst[i + 2] = (bf16_t)f.z;
    dst[i + 3] = (bf16_t)f.w;
  }
}

// ------------------------------------------------- QKV projection (x @ W^T + b)
// grid: (kM/128, kH/128, 3). 256 thr = 8 waves; wave (wm,wn) owns 32x64 of the
// 128x128 block tile (2x4 WMMA tiles). K-step = 32. Tiles fed by TDM.
__global__ __launch_bounds__(256) void gemm_qkv_kernel(
    const bf16_t* __restrict__ xh,
    const bf16_t* __restrict__ Wqh, const bf16_t* __restrict__ Wkh, const bf16_t* __restrict__ Wvh,
    const float* __restrict__ bq, const float* __restrict__ bk, const float* __restrict__ bv,
    bf16_t* __restrict__ Qh, bf16_t* __restrict__ Kh, bf16_t* __restrict__ Vth) {
  __shared__ __align__(16) bf16_t As[128 * 32];
  __shared__ __align__(16) bf16_t Bs[128 * 32];

  const int z = blockIdx.z;
  const bf16_t* Wh  = (z == 0) ? Wqh : (z == 1) ? Wkh : Wvh;
  const float* bias = (z == 0) ? bq  : (z == 1) ? bk  : bv;

  const int m0 = blockIdx.x * 128;
  const int n0 = blockIdx.y * 128;
  const int tid = threadIdx.x;
  const int lane = tid & 31;
  const int w = tid >> 5;
  const int wm = w >> 1;  // 0..3
  const int wn = w & 1;   // 0..1

  const unsigned int ldsA = (unsigned int)(uintptr_t)(&As[0]);
  const unsigned int ldsB = (unsigned int)(uintptr_t)(&Bs[0]);

  v8f acc[2][4] = {};

  for (int kt = 0; kt < kH / 32; ++kt) {
    if (w == 0) {
      tdm_load_tile_2d_bf16(ldsA, xh + (size_t)m0 * kH + kt * 32, 32, 128, kH);
      tdm_load_tile_2d_bf16(ldsB, Wh + (size_t)n0 * kH + kt * 32, 32, 128, kH);
      __builtin_amdgcn_s_wait_tensorcnt(0);
    }
    __syncthreads();

    v16bf aF[2], bF[4];
#pragma unroll
    for (int mi = 0; mi < 2; ++mi)
      aF[mi] = load_fragA_s(&As[(wm * 32 + mi * 16) * 32], lane, 32);
#pragma unroll
    for (int ni = 0; ni < 4; ++ni)
      bF[ni] = load_fragB_s(&Bs[(wn * 64 + ni * 16) * 32], lane, 32);
#pragma unroll
    for (int mi = 0; mi < 2; ++mi)
#pragma unroll
      for (int ni = 0; ni < 4; ++ni)
        acc[mi][ni] = __builtin_amdgcn_wmma_f32_16x16x32_bf16(
            false, aF[mi], false, bF[ni], (short)0, acc[mi][ni], false, false);
    __syncthreads();
  }

  // Epilogue: bias add, scatter to [b,nh,s,d] (Q,K) / transposed [b,nh,d,s] (V)
  const int ln = lane & 15, hi = lane >> 4;
#pragma unroll
  for (int mi = 0; mi < 2; ++mi)
#pragma unroll
    for (int ni = 0; ni < 4; ++ni)
#pragma unroll
      for (int r = 0; r < 8; ++r) {
        int m = m0 + wm * 32 + mi * 16 + r + 8 * hi;
        int n = n0 + wn * 64 + ni * 16 + ln;
        float v = acc[mi][ni][r] + bias[n];
        int b = m >> 11, s = m & (kS - 1);
        int nh = n >> 6, d = n & 63;
        size_t bh = (size_t)(b * kNH + nh);
        bf16_t h = (bf16_t)v;
        if (z == 0)      Qh[(bh * kS + s) * kHD + d] = h;
        else if (z == 1) Kh[(bh * kS + s) * kHD + d] = h;
        else             Vth[(bh * kHD + d) * kS + s] = h;
      }
}

// ---------------------------------------------------- flash attention streaming
// grid: (kS/128, kB*kNH). 8 waves; each wave owns a 16-row query strip.
__global__ __launch_bounds__(256) void attn_kernel(
    const bf16_t* __restrict__ Qh, const bf16_t* __restrict__ Kh,
    const bf16_t* __restrict__ Vth, bf16_t* __restrict__ Oh) {
  __shared__ __align__(16) bf16_t Ks[32 * 64];      // [key][d]
  __shared__ __align__(16) bf16_t Vst[64 * 32];     // [d][key] (from transposed V)
  __shared__ __align__(16) bf16_t Ps[8 * 16 * 32];  // wave-private P tiles

  const int bh = blockIdx.y;
  const int qt = blockIdx.x;
  const int tid = threadIdx.x, lane = tid & 31, w = tid >> 5;
  const int ln = lane & 15, hi = lane >> 4;

  const bf16_t* Qbh = Qh + (size_t)bh * kS * kHD;
  const bf16_t* Kbh = Kh + (size_t)bh * kS * kHD;
  const bf16_t* Vbh = Vth + (size_t)bh * kHD * kS;

  const int qrow0 = qt * 128 + w * 16;

  // Q fragments (16 rows x 64 k) resident in registers for the whole pass
  v16bf qf0 = load_fragA_s(Qbh + (size_t)qrow0 * kHD + 0, lane, kHD);
  v16bf qf1 = load_fragA_s(Qbh + (size_t)qrow0 * kHD + 32, lane, kHD);

  v8f o[4] = {};
  float m_run[8], l_run[8];
#pragma unroll
  for (int r = 0; r < 8; ++r) { m_run[r] = -3.0e38f; l_run[r] = 0.f; }

  bf16_t* Pw = &Ps[w * 16 * 32];
  const int krow = tid >> 3, kc = tid & 7;  // K tile: 32 rows x 8 chunks
  const int vrow = tid >> 2, vc = tid & 3;  // V tile: 64 d-rows x 4 chunks

  for (int kt = 0; kt < kS / 32; ++kt) {
    *(uint4*)&Ks[krow * 64 + kc * 8] =
        *(const uint4*)(Kbh + (size_t)(kt * 32 + krow) * kHD + kc * 8);
    *(uint4*)&Vst[vrow * 32 + vc * 8] =
        *(const uint4*)(Vbh + (size_t)vrow * kS + kt * 32 + vc * 8);
    if (kt + 1 < kS / 32) {
      __builtin_prefetch(Kbh + (size_t)((kt + 1) * 32 + krow) * kHD + kc * 8, 0, 1);
      __builtin_prefetch(Vbh + (size_t)vrow * kS + (kt + 1) * 32 + vc * 8, 0, 1);
    }
    __syncthreads();

    // S = Q K^T : two 16x16 score tiles, K-dim 64 = 2 WMMA steps each
    v8f sc[2] = {};
#pragma unroll
    for (int nt = 0; nt < 2; ++nt) {
      v16bf b0 = load_fragB_s(&Ks[(nt * 16) * 64 + 0], lane, 64);
      v16bf b1 = load_fragB_s(&Ks[(nt * 16) * 64 + 32], lane, 64);
      sc[nt] = __builtin_amdgcn_wmma_f32_16x16x32_bf16(false, qf0, false, b0, (short)0, sc[nt], false, false);
      sc[nt] = __builtin_amdgcn_wmma_f32_16x16x32_bf16(false, qf1, false, b1, (short)0, sc[nt], false, false);
    }

    // online softmax update (rows live in (r,hi); 16 lanes share a row)
#pragma unroll
    for (int r = 0; r < 8; ++r) {
      float s0 = sc[0][r] * 0.125f;  // 1/sqrt(64)
      float s1 = sc[1][r] * 0.125f;
      float mx = fmaxf(s0, s1);
      mx = fmaxf(mx, __shfl_xor(mx, 1));
      mx = fmaxf(mx, __shfl_xor(mx, 2));
      mx = fmaxf(mx, __shfl_xor(mx, 4));
      mx = fmaxf(mx, __shfl_xor(mx, 8));
      float mnew = fmaxf(m_run[r], mx);
      float al = __expf(m_run[r] - mnew);
      float p0 = __expf(s0 - mnew);
      float p1 = __expf(s1 - mnew);
      float rs = p0 + p1;
      rs += __shfl_xor(rs, 1);
      rs += __shfl_xor(rs, 2);
      rs += __shfl_xor(rs, 4);
      rs += __shfl_xor(rs, 8);
      l_run[r] = l_run[r] * al + rs;
      m_run[r] = mnew;
      o[0][r] *= al; o[1][r] *= al; o[2][r] *= al; o[3][r] *= al;
      int row = r + 8 * hi;
      Pw[row * 32 + ln]      = (bf16_t)p0;
      Pw[row * 32 + ln + 16] = (bf16_t)p1;
    }
    __syncthreads();

    // O += P V : P (16x32) in A layout via LDS, V^T tile gives clean B frags
    v16bf aP = load_fragA_s(Pw, lane, 32);
#pragma unroll
    for (int dt = 0; dt < 4; ++dt) {
      v16bf bV = load_fragB_s(&Vst[(dt * 16) * 32], lane, 32);
      o[dt] = __builtin_amdgcn_wmma_f32_16x16x32_bf16(false, aP, false, bV, (short)0, o[dt], false, false);
    }
    __syncthreads();
  }

  // normalize and write attended as [b, s, nh*64 + d] bf16
  const int nh = bh & (kNH - 1);
  const int b  = bh >> 4;
#pragma unroll
  for (int dt = 0; dt < 4; ++dt)
#pragma unroll
    for (int r = 0; r < 8; ++r) {
      int s = qrow0 + r + 8 * hi;
      int d = dt * 16 + ln;
      float val = o[dt][r] / l_run[r];
      Oh[((size_t)(b * kS + s)) * kH + nh * kHD + d] = (bf16_t)val;
    }
}

// ---------------------------------------- out-proj + bias + residual (f32 out)
__global__ __launch_bounds__(256) void gemm_out_kernel(
    const bf16_t* __restrict__ Ah, const bf16_t* __restrict__ Woh,
    const float* __restrict__ bo, const float* __restrict__ xf,
    float* __restrict__ y) {
  __shared__ __align__(16) bf16_t As[128 * 32];
  __shared__ __align__(16) bf16_t Bs[128 * 32];

  const int m0 = blockIdx.x * 128;
  const int n0 = blockIdx.y * 128;
  const int tid = threadIdx.x;
  const int lane = tid & 31;
  const int w = tid >> 5;
  const int wm = w >> 1;
  const int wn = w & 1;

  const unsigned int ldsA = (unsigned int)(uintptr_t)(&As[0]);
  const unsigned int ldsB = (unsigned int)(uintptr_t)(&Bs[0]);

  v8f acc[2][4] = {};

  for (int kt = 0; kt < kH / 32; ++kt) {
    if (w == 0) {
      tdm_load_tile_2d_bf16(ldsA, Ah + (size_t)m0 * kH + kt * 32, 32, 128, kH);
      tdm_load_tile_2d_bf16(ldsB, Woh + (size_t)n0 * kH + kt * 32, 32, 128, kH);
      __builtin_amdgcn_s_wait_tensorcnt(0);
    }
    __syncthreads();

    v16bf aF[2], bF[4];
#pragma unroll
    for (int mi = 0; mi < 2; ++mi)
      aF[mi] = load_fragA_s(&As[(wm * 32 + mi * 16) * 32], lane, 32);
#pragma unroll
    for (int ni = 0; ni < 4; ++ni)
      bF[ni] = load_fragB_s(&Bs[(wn * 64 + ni * 16) * 32], lane, 32);
#pragma unroll
    for (int mi = 0; mi < 2; ++mi)
#pragma unroll
      for (int ni = 0; ni < 4; ++ni)
        acc[mi][ni] = __builtin_amdgcn_wmma_f32_16x16x32_bf16(
            false, aF[mi], false, bF[ni], (short)0, acc[mi][ni], false, false);
    __syncthreads();
  }

  const int ln = lane & 15, hi = lane >> 4;
#pragma unroll
  for (int mi = 0; mi < 2; ++mi)
#pragma unroll
    for (int ni = 0; ni < 4; ++ni)
#pragma unroll
      for (int r = 0; r < 8; ++r) {
        int m = m0 + wm * 32 + mi * 16 + r + 8 * hi;
        int n = n0 + wn * 64 + ni * 16 + ln;
        size_t idx = (size_t)m * kH + n;
        y[idx] = acc[mi][ni][r] + bo[n] + xf[idx];
      }
}

// ------------------------------------------------------------------ layernorm
__global__ __launch_bounds__(256) void ln_kernel(
    const float* __restrict__ y, const float* __restrict__ gamma,
    const float* __restrict__ beta, float* __restrict__ out) {
  __shared__ float red[256];
  __shared__ float red2[256];
  const int row = blockIdx.x;
  const int t = threadIdx.x;
  const float* yr = y + (size_t)row * kH;
  float v[4];
  float s = 0.f, s2 = 0.f;
#pragma unroll
  for (int j = 0; j < 4; ++j) {
    v[j] = yr[t + j * 256];
    s += v[j];
    s2 += v[j] * v[j];
  }
  red[t] = s;
  red2[t] = s2;
  __syncthreads();
  for (int off = 128; off > 0; off >>= 1) {
    if (t < off) { red[t] += red[t + off]; red2[t] += red2[t + off]; }
    __syncthreads();
  }
  float mu = red[0] * (1.f / kH);
  float var = red2[0] * (1.f / kH) - mu * mu;
  float rstd = rsqrtf(var + 1e-5f);
  float* orow = out + (size_t)row * kH;
#pragma unroll
  for (int j = 0; j < 4; ++j) {
    int c = t + j * 256;
    orow[c] = (v[j] - mu) * rstd * gamma[c] + beta[c];
  }
}

extern "C" void kernel_launch(void* const* d_in, const int* in_sizes, int n_in,
                              void* d_out, int out_size, void* d_ws, size_t ws_size,
                              hipStream_t stream) {
  const float* x  = (const float*)d_in[0];
  const float* Wq = (const float*)d_in[1];
  const float* bq = (const float*)d_in[2];
  const float* Wk = (const float*)d_in[3];
  const float* bk = (const float*)d_in[4];
  const float* Wv = (const float*)d_in[5];
  const float* bv = (const float*)d_in[6];
  const float* Wo = (const float*)d_in[7];
  const float* bo = (const float*)d_in[8];
  const float* g  = (const float*)d_in[9];
  const float* be = (const float*)d_in[10];
  float* out = (float*)d_out;

  char* w = (char*)d_ws;
  const size_t MB = 1024u * 1024u;
  bf16_t* xh  = (bf16_t*)(w + 0 * MB);   // 8 MB  [4096,1024]
  bf16_t* Wqh = (bf16_t*)(w + 8 * MB);   // 2 MB
  bf16_t* Wkh = (bf16_t*)(w + 10 * MB);  // 2 MB
  bf16_t* Wvh = (bf16_t*)(w + 12 * MB);  // 2 MB
  bf16_t* Woh = (bf16_t*)(w + 14 * MB);  // 2 MB
  bf16_t* Qh  = (bf16_t*)(w + 16 * MB);  // 8 MB  [bh, s, d]
  bf16_t* Kh  = (bf16_t*)(w + 24 * MB);  // 8 MB  [bh, s, d]
  bf16_t* Vth = (bf16_t*)(w + 32 * MB);  // 8 MB  [bh, d, s]  (transposed)
  bf16_t* Ah  = (bf16_t*)(w + 40 * MB);  // 8 MB  [4096,1024] attended
  float*  y   = (float*)(w + 48 * MB);   // 16 MB residual pre-LN

  cvt_f32_bf16_kernel<<<(kM * kH) / 1024, 256, 0, stream>>>(x, xh, kM * kH);
  cvt_f32_bf16_kernel<<<(kH * kH) / 1024, 256, 0, stream>>>(Wq, Wqh, kH * kH);
  cvt_f32_bf16_kernel<<<(kH * kH) / 1024, 256, 0, stream>>>(Wk, Wkh, kH * kH);
  cvt_f32_bf16_kernel<<<(kH * kH) / 1024, 256, 0, stream>>>(Wv, Wvh, kH * kH);
  cvt_f32_bf16_kernel<<<(kH * kH) / 1024, 256, 0, stream>>>(Wo, Woh, kH * kH);

  gemm_qkv_kernel<<<dim3(kM / 128, kH / 128, 3), 256, 0, stream>>>(
      xh, Wqh, Wkh, Wvh, bq, bk, bv, Qh, Kh, Vth);

  attn_kernel<<<dim3(kS / 128, kB * kNH), 256, 0, stream>>>(Qh, Kh, Vth, Ah);

  gemm_out_kernel<<<dim3(kM / 128, kH / 128), 256, 0, stream>>>(Ah, Woh, bo, x, y);

  ln_kernel<<<kM, 256, 0, stream>>>(y, g, be, out);
}
